// MultiHeadAttention_66967130079590
// MI455X (gfx1250) — matrix operand order repaired and checked by reading, not compiled, not run on hardware
//
#include <hip/hip_runtime.h>
#include <hip/hip_bf16.h>

// Problem constants (B=2, S=2048, D=1024, H=16, hd=64)
#define HH 16
#define BB 2
#define SS 2048
#define DD 1024
#define HD 64
#define NROWS (BB * SS)

typedef __attribute__((ext_vector_type(16))) __bf16 v16bf;
typedef __attribute__((ext_vector_type(8)))  float  v8f;
typedef __attribute__((ext_vector_type(4)))  unsigned int v4u;
typedef __attribute__((ext_vector_type(8)))  int v8i;
typedef __attribute__((ext_vector_type(4)))  int v4i;

#define HAS_TDM __has_builtin(__builtin_amdgcn_tensor_load_to_lds)

union ABfrag { uint4 q[2]; v16bf v; };

__device__ __forceinline__ unsigned short f2bf(float f) {
    unsigned u = __float_as_uint(f);
    u += 0x7fffu + ((u >> 16) & 1u);   // round-to-nearest-even
    return (unsigned short)(u >> 16);
}

__device__ __forceinline__ v8f wmma_bf16(v16bf a, v16bf b, v8f c) {
    return __builtin_amdgcn_wmma_f32_16x16x32_bf16(false, a, false, b,
                                                   (short)0, c, false, false);
}

#if HAS_TDM
// ---------------------------------------------------------------------------
// Issue one 2D TDM tile load (bf16 elements) global -> LDS with row padding.
//   tile0   : contiguous elements per row (X)
//   tile1   : number of rows (Y)
//   stride0 : row stride in elements
//   padi    : pad_interval code (2^(padi+1) DWORDs between pads)
//   pada    : pad_amount code (pada+1 DWORDs inserted)
// ---------------------------------------------------------------------------
__device__ __forceinline__ void tdm_load_2d(unsigned lds_off, const void* gptr,
                                            unsigned tile0, unsigned tile1,
                                            unsigned long long stride0,
                                            unsigned padi, unsigned pada) {
    unsigned long long ga = (unsigned long long)gptr;
    v4u g0;
    g0.x = 1u;                                            // count=1
    g0.y = lds_off;                                       // lds_addr (bytes)
    g0.z = (unsigned)(ga & 0xffffffffull);                // global_addr[31:0]
    g0.w = (unsigned)((ga >> 32) & 0x1ffffffull) | (2u << 30); // addr[56:32], type=2
    v8i g1;
    unsigned dw0 = (1u << 16)                             // data_size = 1 -> 2 bytes
                 | (1u << 20)                             // pad_enable
                 | (padi << 22) | (pada << 25);
    g1[0] = (int)dw0;
    g1[1] = (int)((tile0 & 0xffffu) << 16);               // tensor_dim0[15:0]
    g1[2] = (int)(((tile0 >> 16) & 0xffffu) | ((tile1 & 0xffffu) << 16)); // dim0 hi | dim1 lo
    g1[3] = (int)(((tile1 >> 16) & 0xffffu) | ((tile0 & 0xffffu) << 16)); // dim1 hi | tile_dim0
    g1[4] = (int)(tile1 & 0xffffu);                       // tile_dim1 (tile_dim2 = 0)
    g1[5] = (int)(unsigned)(stride0 & 0xffffffffull);     // tensor_dim0_stride lo
    g1[6] = (int)(unsigned)((stride0 >> 32) & 0xffffull); // stride hi (dim1_stride = 0)
    g1[7] = 0;
    v4i gz = {0, 0, 0, 0};
#if __clang_major__ >= 23
    v8i z8 = {0, 0, 0, 0, 0, 0, 0, 0};
    __builtin_amdgcn_tensor_load_to_lds(g0, g1, gz, gz, z8, 0);
#else
    __builtin_amdgcn_tensor_load_to_lds(g0, g1, gz, gz, 0);
#endif
}
#endif  // HAS_TDM

// ---------------------------------------------------------------------------
// Convert x f32 -> bf16 (8 elems / thread)
// ---------------------------------------------------------------------------
__global__ __launch_bounds__(256) void k_cvt_x(const float* __restrict__ x,
                                               unsigned short* __restrict__ xb) {
    int t = blockIdx.x * blockDim.x + threadIdx.x;
    const float4* p = (const float4*)x + (size_t)t * 2;
    float4 a = p[0], b = p[1];
    uint4 o;
    o.x = (unsigned)f2bf(a.x) | ((unsigned)f2bf(a.y) << 16);
    o.y = (unsigned)f2bf(a.z) | ((unsigned)f2bf(a.w) << 16);
    o.z = (unsigned)f2bf(b.x) | ((unsigned)f2bf(b.y) << 16);
    o.w = (unsigned)f2bf(b.z) | ((unsigned)f2bf(b.w) << 16);
    *(uint4*)(xb + (size_t)t * 8) = o;
}

// ---------------------------------------------------------------------------
// Pack weights: bf16, transposed, interleaved-head permutation folded in.
// ---------------------------------------------------------------------------
__global__ __launch_bounds__(256) void k_pack_w(const float* __restrict__ qw,
                                                const float* __restrict__ kw,
                                                const float* __restrict__ vw,
                                                const float* __restrict__ ow,
                                                unsigned short* __restrict__ wqT,
                                                unsigned short* __restrict__ wkT,
                                                unsigned short* __restrict__ wvT,
                                                unsigned short* __restrict__ woT) {
    int t   = blockIdx.x * blockDim.x + threadIdx.x;
    int mat = t >> 20;
    int r   = (t >> 10) & 1023;
    int c   = t & 1023;
    if (mat < 3) {
        const float* w = (mat == 0) ? qw : (mat == 1) ? kw : vw;
        unsigned short* dst = (mat == 0) ? wqT : (mat == 1) ? wkT : wvT;
        dst[r * DD + c] = f2bf(w[c * DD + ((r & 63) * 16 + (r >> 6))]);
    } else {
        woT[r * DD + c] = f2bf(ow[((c & 63) * 16 + (c >> 6)) * DD + r]);
    }
}

// ---------------------------------------------------------------------------
// bf16 GEMM, TDM-fed + double-buffered LDS. Tile 128x128x32, 8 waves,
// wave = 64x32 = 8 WMMA frags. Bt stored [n][k] (pre-transposed weights).
// Epilogue modes: 0 = f32 row-major, 1 = bf16 [bh][s][c], 3 = bf16 [bh][c][s].
// ---------------------------------------------------------------------------
__global__ __launch_bounds__(256) void k_gemm(const unsigned short* __restrict__ A,
                                              const unsigned short* __restrict__ Bt,
                                              float* __restrict__ outF,
                                              unsigned short* __restrict__ outB,
                                              int mode, float scale) {
    __shared__ __align__(16) unsigned short At[2][128 * 40];
    __shared__ __align__(16) unsigned short Bl[2][128 * 40];

    const int tid = threadIdx.x;
    const int wid = tid >> 5, lane = tid & 31, ln = lane & 15, hi = lane >> 4;
    const int m0 = (wid >> 2) * 64, n0 = (wid & 3) * 32;
    const int bm = blockIdx.y * 128, bn = blockIdx.x * 128;

    v8f acc[4][2];
    const v8f z = {0.f, 0.f, 0.f, 0.f, 0.f, 0.f, 0.f, 0.f};
    for (int i = 0; i < 4; ++i)
        for (int j = 0; j < 2; ++j) acc[i][j] = z;

    const unsigned short* gA = A  + (size_t)bm * DD;
    const unsigned short* gB = Bt + (size_t)bn * DD;
    const int NIT = DD / 32;

#if HAS_TDM
    if (wid == 0) {   // preload tile 0 (32-elem rows, +8 pad -> pitch 40)
        tdm_load_2d((unsigned)(unsigned long long)&At[0][0], gA, 32, 128, DD, 3, 3);
        tdm_load_2d((unsigned)(unsigned long long)&Bl[0][0], gB, 32, 128, DD, 3, 3);
    }
#endif

    for (int it = 0; it < NIT; ++it) {
        const int cur = it & 1, nxt = cur ^ 1;
        __syncthreads();   // buffer[nxt] readers (iter it-1) are done
#if HAS_TDM
        if (wid == 0) {
            if (it + 1 < NIT) {
                tdm_load_2d((unsigned)(unsigned long long)&At[nxt][0],
                            gA + (it + 1) * 32, 32, 128, DD, 3, 3);
                tdm_load_2d((unsigned)(unsigned long long)&Bl[nxt][0],
                            gB + (it + 1) * 32, 32, 128, DD, 3, 3);
                __builtin_amdgcn_s_wait_tensorcnt(2);  // in-order: current pair done
            } else {
                __builtin_amdgcn_s_wait_tensorcnt(0);
            }
        }
#else
        {   // manual staging fallback
            const int lrow = tid >> 1, lhalf = (tid & 1) * 16;
            const unsigned short* ga = gA + (size_t)lrow * DD + it * 32 + lhalf;
            const unsigned short* gb = gB + (size_t)lrow * DD + it * 32 + lhalf;
            uint4 a0 = *(const uint4*)ga, a1 = *(const uint4*)(ga + 8);
            uint4 b0 = *(const uint4*)gb, b1 = *(const uint4*)(gb + 8);
            *(uint4*)&At[cur][lrow * 40 + lhalf]     = a0;
            *(uint4*)&At[cur][lrow * 40 + lhalf + 8] = a1;
            *(uint4*)&Bl[cur][lrow * 40 + lhalf]     = b0;
            *(uint4*)&Bl[cur][lrow * 40 + lhalf + 8] = b1;
        }
#endif
        __syncthreads();   // tile[cur] visible

        ABfrag af[4], bf[2];
        const int klo = hi * 8;
        for (int mt = 0; mt < 4; ++mt) {
            int row = m0 + mt * 16 + ln;
            af[mt].q[0] = *(const uint4*)&At[cur][row * 40 + klo];
            af[mt].q[1] = *(const uint4*)&At[cur][row * 40 + klo + 16];
        }
        const int kb = hi * 8;
        for (int nt = 0; nt < 2; ++nt) {
            int row = n0 + nt * 16 + ln;
            bf[nt].q[0] = *(const uint4*)&Bl[cur][row * 40 + kb];
            bf[nt].q[1] = *(const uint4*)&Bl[cur][row * 40 + kb + 16];
        }
        for (int mt = 0; mt < 4; ++mt)
            for (int nt = 0; nt < 2; ++nt)
                acc[mt][nt] = wmma_bf16(af[mt].v, bf[nt].v, acc[mt][nt]);
    }

    for (int mt = 0; mt < 4; ++mt)
        for (int nt = 0; nt < 2; ++nt)
            for (int r = 0; r < 8; ++r) {
                int i = bm + m0 + mt * 16 + hi * 8 + r;
                int j = bn + n0 + nt * 16 + ln;
                float v = acc[mt][nt][r] * scale;
                if (mode == 0) {
                    outF[(size_t)i * DD + j] = v;
                } else {
                    int b = i >> 11, s = i & 2047, h = j >> 6, c = j & 63;
                    size_t idx = (mode == 3)
                        ? ((size_t)((b * HH + h) * HD + c) * SS + s)
                        : ((size_t)((b * HH + h) * SS + s) * HD + c);
                    outB[idx] = f2bf(v);
                }
            }
}

// ---------------------------------------------------------------------------
// Causal flash attention, TDM-fed + double-buffered K/V tiles.
// Block: 256 thr = 8 waves, 128 q-rows; wave owns 16 q-rows x 64 features.
// ---------------------------------------------------------------------------
__global__ __launch_bounds__(256) void k_attn(const unsigned short* __restrict__ Q,
                                              const unsigned short* __restrict__ K,
                                              const unsigned short* __restrict__ Vt,
                                              unsigned short* __restrict__ AO) {
    __shared__ __align__(16) unsigned short Kl[2][64 * 72];     // [key][feat]
    __shared__ __align__(16) unsigned short Vl[2][64 * 72];     // [feat][key]
    __shared__ __align__(16) unsigned short Pl[8][16 * 72];     // per-wave P

    const int tid = threadIdx.x, wid = tid >> 5, lane = tid & 31;
    const int ln = lane & 15, hi = lane >> 4;
    const int bh = blockIdx.y, b = bh >> 4, h = bh & 15;
    const int qbase = blockIdx.x * 128;
    const size_t qkBase = (size_t)bh * SS * HD;
    const size_t vBase  = (size_t)bh * HD * SS;

    ABfrag qa[2];
    {
        const unsigned short* qrow = Q + qkBase + (size_t)(qbase + wid * 16 + ln) * HD;
        int klo = hi * 8;
        qa[0].q[0] = *(const uint4*)(qrow + klo);
        qa[0].q[1] = *(const uint4*)(qrow + klo + 16);
        qa[1].q[0] = *(const uint4*)(qrow + 32 + klo);
        qa[1].q[1] = *(const uint4*)(qrow + 32 + klo + 16);
    }

    const v8f z = {0.f, 0.f, 0.f, 0.f, 0.f, 0.f, 0.f, 0.f};
    v8f o[4];
    float mrow[8], lrow[8];
    for (int nt = 0; nt < 4; ++nt) o[nt] = z;
    for (int r = 0; r < 8; ++r) { mrow[r] = -1e30f; lrow[r] = 0.f; }

    const int nkb = blockIdx.x * 2 + 2;   // causal: keys < qbase + 128

#if HAS_TDM
    if (wid == 0) {   // preload K/V tile 0 (64-elem rows, +8 pad -> pitch 72)
        tdm_load_2d((unsigned)(unsigned long long)&Kl[0][0],
                    K + qkBase, 64, 64, HD, 4, 3);
        tdm_load_2d((unsigned)(unsigned long long)&Vl[0][0],
                    Vt + vBase, 64, 64, SS, 4, 3);
    }
#endif

    for (int kb = 0; kb < nkb; ++kb) {
        const int cur = kb & 1, nxt = cur ^ 1;
        __syncthreads();   // buffer[nxt] readers (iter kb-1) done
#if HAS_TDM
        if (wid == 0) {
            if (kb + 1 < nkb) {
                tdm_load_2d((unsigned)(unsigned long long)&Kl[nxt][0],
                            K + qkBase + (size_t)(kb + 1) * 64 * HD, 64, 64, HD, 4, 3);
                tdm_load_2d((unsigned)(unsigned long long)&Vl[nxt][0],
                            Vt + vBase + (size_t)(kb + 1) * 64, 64, 64, SS, 4, 3);
                __builtin_amdgcn_s_wait_tensorcnt(2);
            } else {
                __builtin_amdgcn_s_wait_tensorcnt(0);
            }
        }
#else
        {   // manual staging fallback
            const int srow = tid >> 2, squad = (tid & 3) * 16;
            const unsigned short* kg = K  + qkBase + (size_t)(kb * 64 + srow) * HD + squad;
            const unsigned short* vg = Vt + vBase  + (size_t)srow * SS + kb * 64 + squad;
            uint4 kv0 = *(const uint4*)kg, kv1 = *(const uint4*)(kg + 8);
            uint4 vv0 = *(const uint4*)vg, vv1 = *(const uint4*)(vg + 8);
            *(uint4*)&Kl[cur][srow * 72 + squad]     = kv0;
            *(uint4*)&Kl[cur][srow * 72 + squad + 8] = kv1;
            *(uint4*)&Vl[cur][srow * 72 + squad]     = vv0;
            *(uint4*)&Vl[cur][srow * 72 + squad + 8] = vv1;
        }
#endif
        __syncthreads();   // tile[cur] visible

        // scores S = Q * K^T
        v8f sf[4];
        for (int nt = 0; nt < 4; ++nt) {
            v8f s = z;
            for (int ks = 0; ks < 2; ++ks) {
                ABfrag bfr;
                int row = nt * 16 + ln;
                int kk  = ks * 32 + hi * 8;
                bfr.q[0] = *(const uint4*)&Kl[cur][row * 72 + kk];
                bfr.q[1] = *(const uint4*)&Kl[cur][row * 72 + kk + 16];
                s = wmma_bf16(qa[ks].v, bfr.v, s);
            }
            sf[nt] = s;
        }

        // online softmax (C layout: lane = key col, VGPR r = row r + 8*hi)
        for (int r = 0; r < 8; ++r) {
            int qidx = qbase + wid * 16 + hi * 8 + r;
            float vals[4];
            float mx = mrow[r];
            for (int nt = 0; nt < 4; ++nt) {
                int kidx = kb * 64 + nt * 16 + ln;
                vals[nt] = (kidx <= qidx) ? sf[nt][r] : -1e30f;
                mx = fmaxf(mx, vals[nt]);
            }
            for (int off = 1; off < 16; off <<= 1)
                mx = fmaxf(mx, __shfl_xor(mx, off, 32));
            float alpha = __expf(mrow[r] - mx);
            float rs = 0.f;
            for (int nt = 0; nt < 4; ++nt) {
                float p = (vals[nt] > -5e29f) ? __expf(vals[nt] - mx) : 0.f;
                rs += p;
                Pl[wid][(hi * 8 + r) * 72 + nt * 16 + ln] = f2bf(p);
            }
            for (int off = 1; off < 16; off <<= 1)
                rs += __shfl_xor(rs, off, 32);
            lrow[r] = lrow[r] * alpha + rs;
            mrow[r] = mx;
            for (int nt = 0; nt < 4; ++nt) o[nt][r] *= alpha;
        }
        __syncthreads();   // P visible within workgroup

        // O += P * V
        for (int ks = 0; ks < 2; ++ks) {
            ABfrag pa;
            int klo = hi * 8;
            pa.q[0] = *(const uint4*)&Pl[wid][ln * 72 + ks * 32 + klo];
            pa.q[1] = *(const uint4*)&Pl[wid][ln * 72 + ks * 32 + klo + 16];
            for (int nt = 0; nt < 4; ++nt) {
                ABfrag bv;
                int row = nt * 16 + ln;
                int kk  = ks * 32 + hi * 8;
                bv.q[0] = *(const uint4*)&Vl[cur][row * 72 + kk];
                bv.q[1] = *(const uint4*)&Vl[cur][row * 72 + kk + 16];
                o[nt] = wmma_bf16(pa.v, bv.v, o[nt]);
            }
        }
    }

    for (int r = 0; r < 8; ++r) {
        float inv = 1.0f / lrow[r];
        int srow_q = qbase + wid * 16 + hi * 8 + r;
        size_t obase = (size_t)(b * SS + srow_q) * DD + h * HD;
        for (int nt = 0; nt < 4; ++nt)
            AO[obase + nt * 16 + ln] = f2bf(o[nt][r] * inv);
    }
}

// ---------------------------------------------------------------------------
extern "C" void kernel_launch(void* const* d_in, const int* in_sizes, int n_in,
                              void* d_out, int out_size, void* d_ws, size_t ws_size,
                              hipStream_t stream) {
    (void)in_sizes; (void)n_in; (void)out_size; (void)ws_size;
    const float* x  = (const float*)d_in[0];
    const float* qw = (const float*)d_in[1];
    const float* kw = (const float*)d_in[2];
    const float* vw = (const float*)d_in[3];
    const float* ow = (const float*)d_in[4];
    float* out = (float*)d_out;

    char* w = (char*)d_ws;
    const size_t MiB = 1024 * 1024;
    unsigned short* xb  = (unsigned short*)(w + 0);
    unsigned short* wqT = (unsigned short*)(w + 8  * MiB);
    unsigned short* wkT = (unsigned short*)(w + 10 * MiB);
    unsigned short* wvT = (unsigned short*)(w + 12 * MiB);
    unsigned short* woT = (unsigned short*)(w + 14 * MiB);
    unsigned short* Qb  = (unsigned short*)(w + 16 * MiB);
    unsigned short* Kb  = (unsigned short*)(w + 24 * MiB);
    unsigned short* Vtb = (unsigned short*)(w + 32 * MiB);
    unsigned short* AOb = (unsigned short*)(w + 40 * MiB);

    k_cvt_x<<<(NROWS * DD) / 8 / 256, 256, 0, stream>>>(x, xb);
    k_pack_w<<<(4 * DD * DD) / 256, 256, 0, stream>>>(qw, kw, vw, ow,
                                                      wqT, wkT, wvT, woT);
    dim3 gg(DD / 128, NROWS / 128);
    k_gemm<<<gg, 256, 0, stream>>>(xb, wqT, nullptr, Qb,  1, 0.03125f);
    k_gemm<<<gg, 256, 0, stream>>>(xb, wkT, nullptr, Kb,  1, 1.0f);
    k_gemm<<<gg, 256, 0, stream>>>(xb, wvT, nullptr, Vtb, 3, 1.0f);
    k_attn<<<dim3(SS / 128, BB * HH), 256, 0, stream>>>(Qb, Kb, Vtb, AOb);
    k_gemm<<<gg, 256, 0, stream>>>(AOb, woT, out, nullptr, 0, 1.0f);
}
